// DecoderRNN_21139829031781
// MI455X (gfx1250) — matrix operand order repaired and checked by reading, not compiled, not run on hardware
//
#include <hip/hip_runtime.h>

// ---------------- types / helpers ----------------
typedef __attribute__((ext_vector_type(16))) __bf16 v16bf;
typedef __attribute__((ext_vector_type(8)))  float  v8f;

static __device__ __forceinline__ unsigned short f2bf(float x) {
    unsigned int u = __builtin_bit_cast(unsigned int, x);
    unsigned int r = u + 0x7FFFu + ((u >> 16) & 1u);   // round-to-nearest-even
    return (unsigned short)(r >> 16);
}
static __device__ __forceinline__ float bf2f(unsigned short u) {
    unsigned int x = ((unsigned int)u) << 16;
    return __builtin_bit_cast(float, x);
}
static __device__ __forceinline__ float sigmoidf_(float x) {
    return 1.0f / (1.0f + __expf(-x));
}

union BV16 { uint4 q[2]; v16bf v; };

// A fragment (16x32 bf16, M=row index = lane&15):
// lanes 0-15 hold K = kk*32 + {0..15}; lanes 16-31 hold K = kk*32 + {16..31}
static __device__ __forceinline__ v16bf load_A16(const unsigned short* row, int kk, int hf) {
    BV16 u;
    const unsigned short* p = row + kk * 32 + hf * 16;
    u.q[0] = *(const uint4*)p;
    u.q[1] = *(const uint4*)(p + 8);
    return u.v;
}
// B fragment: pre-tiled so each lane's 16 bf16 are contiguous (tileBase + lane*16)
static __device__ __forceinline__ v16bf load_B16(const unsigned short* p) {
    BV16 u;
    u.q[0] = *(const uint4*)p;
    u.q[1] = *(const uint4*)(p + 8);
    return u.v;
}
static __device__ __forceinline__ v8f wmma_bf16(v16bf a, v16bf b, v8f c) {
    return __builtin_amdgcn_wmma_f32_16x16x32_bf16(false, a, false, b, (short)0, c, false, false);
}

// ---------------- problem constants ----------------
#define NB      2048
#define TT      100
#define SMH     256
#define EMB     64
#define HID     1024
#define PLEN    45
#define NVOC    100
#define FEAT    512        // 2*SMH

// workspace (bf16 element offsets) — weights pre-tiled to WMMA lane order
#define SZ_WH   (3*64*32*512)          // 3,145,728
#define OFF_WH  0
#define SZ_WI   (3*64*2*512)           // 196,608
#define OFF_WI  (OFF_WH + SZ_WH)
#define SZ_WFC1 (64*16*512)            // 524,288
#define OFF_WFC1 (OFF_WI + SZ_WI)
#define SZ_WOUT (7*32*512)             // 114,688 (N padded 100->112)
#define OFF_WOUT (OFF_WFC1 + SZ_WFC1)
#define SZ_EMB  (NVOC*EMB)             // 6,400  (relu pre-applied)
#define OFF_EMB (OFF_WOUT + SZ_WOUT)
#define WS_TOTAL (OFF_EMB + SZ_EMB)    // 3,987,712 bf16 = ~7.98 MB

// d_out layout (floats): log_probs[2048,100] | rnn_out[2048,45,1024] | decoded[46,2048]
#define OUT_LP   0L
#define OUT_RNN  (2048L*100L)
#define OUT_DEC  (OUT_RNN + 2048L*45L*1024L)

#define PB 1032   // LDS pitch for h (bf16 elems): 2064B -> bank-conflict-free b128 A loads

// ---------------- prologue: convert+tile weights into ws ----------------
__global__ __launch_bounds__(256) void pretile_kernel(
    const float* __restrict__ Wi, const float* __restrict__ Wh,
    const float* __restrict__ Wfc1, const float* __restrict__ Wout,
    const float* __restrict__ emb, unsigned short* __restrict__ ws)
{
    int idx = blockIdx.x * 256 + threadIdx.x;
    if (idx >= WS_TOTAL) return;
    float v;
    if (idx < OFF_WI) {                               // Wh [3072,1024]
        int te = idx & 511, lane = te >> 4, e = te & 15;
        int tile = idx >> 9, kk = tile & 31, jc = tile >> 5;
        int j = (jc >> 6) * 1024 + (jc & 63) * 16 + (lane & 15);
        int k = kk * 32 + (lane >> 4) * 16 + e;
        v = Wh[j * 1024 + k];
    } else if (idx < OFF_WFC1) {                      // Wi [3072,64]
        int s = idx - OFF_WI;
        int te = s & 511, lane = te >> 4, e = te & 15;
        int tile = s >> 9, kk = tile & 1, jc = tile >> 1;
        int j = (jc >> 6) * 1024 + (jc & 63) * 16 + (lane & 15);
        int k = kk * 32 + (lane >> 4) * 16 + e;
        v = Wi[j * 64 + k];
    } else if (idx < OFF_WOUT) {                      // W_fc1 [1024,512]
        int s = idx - OFF_WFC1;
        int te = s & 511, lane = te >> 4, e = te & 15;
        int tile = s >> 9, kk = tile & 15, c = tile >> 4;
        int j = c * 16 + (lane & 15);
        int k = kk * 32 + (lane >> 4) * 16 + e;
        v = Wfc1[j * 512 + k];
    } else if (idx < OFF_EMB) {                       // W_out [100,1024], pad rows>=100 with 0
        int s = idx - OFF_WOUT;
        int te = s & 511, lane = te >> 4, e = te & 15;
        int tile = s >> 9, kk = tile & 31, c = tile >> 5;
        int j = c * 16 + (lane & 15);
        int k = kk * 32 + (lane >> 4) * 16 + e;
        v = (j < NVOC) ? Wout[j * 1024 + k] : 0.0f;
    } else {                                          // relu(emb) [100,64]
        float e = emb[idx - OFF_EMB];
        v = e > 0.0f ? e : 0.0f;
    }
    ws[idx] = f2bf(v);
}

// ---------------- persistent decoder: one block = 16 batch rows, all 45 steps ----------------
__global__ __launch_bounds__(256) void decoder_kernel(
    const float* __restrict__ sm_hidden, const float* __restrict__ b_fc1,
    const float* __restrict__ bi, const float* __restrict__ bh,
    const float* __restrict__ b_out, const unsigned short* __restrict__ ws,
    float* __restrict__ out)
{
    __shared__ __align__(16) unsigned short h_lds[16 * PB];   // h (bf16, A-operand layout source)
    __shared__ __align__(16) float logits_lds[16 * 112];
    __shared__ int tok_lds[16];

    const int tid  = threadIdx.x;
    const int lane = tid & 31;
    const int wave = tid >> 5;
    const int hf   = lane >> 4;     // which half of the wave
    const int lcol = lane & 15;     // N index within tile / M row for A
    const int row0 = blockIdx.x * 16;

    float* out_lp  = out + OUT_LP;
    float* out_rnn = out + OUT_RNN;
    float* out_dec = out + OUT_DEC;

    const unsigned short* Wh_t   = ws + OFF_WH;
    const unsigned short* Wi_t   = ws + OFF_WI;
    const unsigned short* Wfc1_t = ws + OFF_WFC1;
    const unsigned short* Wout_t = ws + OFF_WOUT;
    const unsigned short* emb_b  = ws + OFF_EMB;

    if (tid < 16) { tok_lds[tid] = 0; out_dec[row0 + tid] = 0.0f; }  // sos + decoded[0]

    // ---- feats: concat(max_t, mean_t) over sm_hidden[row, 0:100, 0:256] -> h_lds (bf16) ----
    {
        int m = tid >> 4, g = tid & 15;
        const float* src = sm_hidden + ((long)(row0 + m) * TT) * SMH + g * 16;
        float4 mx[4], sm[4];
        #pragma unroll
        for (int q = 0; q < 4; q++) { float4 v = ((const float4*)src)[q]; mx[q] = v; sm[q] = v; }
        for (int t = 1; t < TT; t++) {
            const float4* p = (const float4*)(src + t * SMH);
            #pragma unroll
            for (int q = 0; q < 4; q++) {
                float4 v = p[q];
                mx[q].x = fmaxf(mx[q].x, v.x); mx[q].y = fmaxf(mx[q].y, v.y);
                mx[q].z = fmaxf(mx[q].z, v.z); mx[q].w = fmaxf(mx[q].w, v.w);
                sm[q].x += v.x; sm[q].y += v.y; sm[q].z += v.z; sm[q].w += v.w;
            }
        }
        unsigned short* dr = &h_lds[m * PB];
        #pragma unroll
        for (int q = 0; q < 4; q++) {
            int c = g * 16 + q * 4;
            dr[c + 0] = f2bf(mx[q].x); dr[c + 1] = f2bf(mx[q].y);
            dr[c + 2] = f2bf(mx[q].z); dr[c + 3] = f2bf(mx[q].w);
            dr[SMH + c + 0] = f2bf(sm[q].x * 0.01f); dr[SMH + c + 1] = f2bf(sm[q].y * 0.01f);
            dr[SMH + c + 2] = f2bf(sm[q].z * 0.01f); dr[SMH + c + 3] = f2bf(sm[q].w * 0.01f);
        }
    }
    __syncthreads();

    unsigned int hpack[8][4];   // held h_new tiles (bf16-packed) across the publish barrier
    const unsigned short* arow = &h_lds[lcol * PB];

    // ---- h0 = relu(feats @ Wfc1^T + b_fc1) ----
    #pragma unroll
    for (int ci = 0; ci < 8; ci++) {
        int c = wave * 8 + ci;
        v8f acc = {0, 0, 0, 0, 0, 0, 0, 0};
        const unsigned short* bp = Wfc1_t + (long)c * 16 * 512 + lane * 16;
        for (int kk = 0; kk < 16; kk++) {
            acc = wmma_bf16(load_A16(arow, kk, hf), load_B16(bp), acc);
            bp += 512;
        }
        float bias = b_fc1[c * 16 + lcol];
        #pragma unroll
        for (int j = 0; j < 4; j++) {
            float v0 = acc[2 * j] + bias;     v0 = v0 > 0.0f ? v0 : 0.0f;
            float v1 = acc[2 * j + 1] + bias; v1 = v1 > 0.0f ? v1 : 0.0f;
            hpack[ci][j] = (unsigned int)f2bf(v0) | ((unsigned int)f2bf(v1) << 16);
        }
    }
    __syncthreads();
    #pragma unroll
    for (int ci = 0; ci < 8; ci++) {
        int c = wave * 8 + ci;
        #pragma unroll
        for (int i = 0; i < 8; i++)
            h_lds[(i + 8 * hf) * PB + c * 16 + lcol] =
                (unsigned short)(hpack[ci][i >> 1] >> (16 * (i & 1)));
    }
    __syncthreads();

    // ---- 45 decode steps ----
    for (int t = 0; t < PLEN; t++) {
        const unsigned short* xrow = emb_b + tok_lds[lcol] * 64;   // relu(emb[tok]) bf16

        #pragma unroll
        for (int ci = 0; ci < 8; ci++) {
            int c = wave * 8 + ci;
            v8f ar  = {0, 0, 0, 0, 0, 0, 0, 0};
            v8f az  = {0, 0, 0, 0, 0, 0, 0, 0};
            v8f ain = {0, 0, 0, 0, 0, 0, 0, 0};
            v8f ahn = {0, 0, 0, 0, 0, 0, 0, 0};
            // gi = x @ Wi^T (K=64)
            {
                const unsigned short* br = Wi_t + (long)((0   + c) * 2) * 512 + lane * 16;
                const unsigned short* bz = Wi_t + (long)((64  + c) * 2) * 512 + lane * 16;
                const unsigned short* bn = Wi_t + (long)((128 + c) * 2) * 512 + lane * 16;
                #pragma unroll
                for (int kk = 0; kk < 2; kk++) {
                    v16bf a = load_A16(xrow, kk, hf);
                    ar  = wmma_bf16(a, load_B16(br + kk * 512), ar);
                    az  = wmma_bf16(a, load_B16(bz + kk * 512), az);
                    ain = wmma_bf16(a, load_B16(bn + kk * 512), ain);
                }
            }
            // gh = h @ Wh^T (K=1024)
            {
                const unsigned short* br = Wh_t + (long)((0   + c) * 32) * 512 + lane * 16;
                const unsigned short* bz = Wh_t + (long)((64  + c) * 32) * 512 + lane * 16;
                const unsigned short* bn = Wh_t + (long)((128 + c) * 32) * 512 + lane * 16;
                for (int kk = 0; kk < 32; kk++) {
                    v16bf a = load_A16(arow, kk, hf);
                    ar  = wmma_bf16(a, load_B16(br), ar);  br += 512;
                    az  = wmma_bf16(a, load_B16(bz), az);  bz += 512;
                    ahn = wmma_bf16(a, load_B16(bn), ahn); bn += 512;
                }
            }
            int col = c * 16 + lcol;
            float brz = bi[col]        + bh[col];
            float bzz = bi[1024 + col] + bh[1024 + col];
            float bin = bi[2048 + col];
            float bhn = bh[2048 + col];
            #pragma unroll
            for (int i = 0; i < 8; i++) {
                int m = i + 8 * hf;
                float hold = bf2f(h_lds[m * PB + col]);            // old h (read before barrier)
                float r = sigmoidf_(ar[i] + brz);
                float z = sigmoidf_(az[i] + bzz);
                float n = ain[i] + bin + r * (ahn[i] + bhn);
                n = n > 0.0f ? n : 0.0f;
                float hn = (1.0f - z) * n + z * hold;
                out_rnn[((long)(row0 + m) * PLEN + t) * HID + col] = hn;
                if ((i & 1) == 0) hpack[ci][i >> 1]  = (unsigned int)f2bf(hn);
                else              hpack[ci][i >> 1] |= ((unsigned int)f2bf(hn) << 16);
            }
        }
        __syncthreads();   // all waves done reading old h
        #pragma unroll
        for (int ci = 0; ci < 8; ci++) {
            int c = wave * 8 + ci;
            #pragma unroll
            for (int i = 0; i < 8; i++)
                h_lds[(i + 8 * hf) * PB + c * 16 + lcol] =
                    (unsigned short)(hpack[ci][i >> 1] >> (16 * (i & 1)));
        }
        __syncthreads();   // h_new published

        // logits = h_new @ W_out^T + b_out  (N padded to 112 -> 7 tiles, waves 0..6)
        if (wave < 7) {
            v8f acc = {0, 0, 0, 0, 0, 0, 0, 0};
            const unsigned short* bp = Wout_t + (long)wave * 32 * 512 + lane * 16;
            for (int kk = 0; kk < 32; kk++) {
                acc = wmma_bf16(load_A16(arow, kk, hf), load_B16(bp), acc);
                bp += 512;
            }
            int col = wave * 16 + lcol;
            float bo = (col < NVOC) ? b_out[col] : 0.0f;
            #pragma unroll
            for (int i = 0; i < 8; i++)
                logits_lds[(i + 8 * hf) * 112 + col] = acc[i] + bo;
        }
        __syncthreads();

        // argmax (+ final log_softmax) — wave 0, one lane per row
        if (wave == 0 && lane < 16) {
            const float* lr = &logits_lds[lane * 112];
            float best = lr[0]; int bidx = 0;
            for (int j = 1; j < NVOC; j++) { float v = lr[j]; if (v > best) { best = v; bidx = j; } }
            tok_lds[lane] = bidx;
            out_dec[(long)(t + 1) * NB + row0 + lane] = (float)bidx;
            if (t == PLEN - 1) {
                float s = 0.0f;
                for (int j = 0; j < NVOC; j++) s += __expf(lr[j] - best);
                float lse = best + __logf(s);
                float* lpr = out_lp + (long)(row0 + lane) * NVOC;
                for (int j = 0; j < NVOC; j++) lpr[j] = lr[j] - lse;
            }
        }
        __syncthreads();
    }
}

extern "C" void kernel_launch(void* const* d_in, const int* in_sizes, int n_in,
                              void* d_out, int out_size, void* d_ws, size_t ws_size,
                              hipStream_t stream) {
    (void)in_sizes; (void)n_in; (void)out_size; (void)ws_size;
    const float* sm_hidden = (const float*)d_in[0];
    const float* emb       = (const float*)d_in[1];
    const float* W_fc1     = (const float*)d_in[2];
    const float* b_fc1     = (const float*)d_in[3];
    const float* Wi        = (const float*)d_in[4];
    const float* Wh        = (const float*)d_in[5];
    const float* bi        = (const float*)d_in[6];
    const float* bh        = (const float*)d_in[7];
    const float* W_out     = (const float*)d_in[8];
    const float* b_out     = (const float*)d_in[9];
    unsigned short* ws     = (unsigned short*)d_ws;
    float* out             = (float*)d_out;

    int ptBlocks = (WS_TOTAL + 255) / 256;
    pretile_kernel<<<ptBlocks, 256, 0, stream>>>(Wi, Wh, W_fc1, W_out, emb, ws);
    decoder_kernel<<<NB / 16, 256, 0, stream>>>(sm_hidden, b_fc1, bi, bh, b_out, ws, out);
}
